// SequentialLlama4TextMoe_55628416418275
// MI455X (gfx1250) — matrix operand order repaired and checked by reading, not compiled, not run on hardware
//
#include <hip/hip_runtime.h>

// ---------------------------------------------------------------------------
// SequentialLlama4TextMoe for MI455X (gfx1250, wave32, WMMA bf16 16x16x32).
// Top-1 routing => compute only the selected expert per token + shared MLP.
// Two-pass MoE MLP (gate/up -> bf16 act in ws, then down) with grouped GEMM.
// Tiles: 32M x 128N per 256-thread block (8 waves), K stepped by 32.
// Pass2 A-tiles staged with GLOBAL_LOAD_ASYNC_TO_LDS_B128 (ASYNCcnt path).
// ws requirement: 64 KB header + 2 * 8 MB bf16 activation buffers (~16.1 MB).
// ---------------------------------------------------------------------------

#define T_TOK 2048
#define H_DIM 1024
#define I_DIM 2048
#define N_EXP 16
#define KSTEP 32          // K per WMMA (bf16)
#define BM    32          // M rows per block (2 x 16 sub-tiles)
#define BN    128         // N columns per block = 8 waves * 16
#define LDSW  40          // LDS row stride in bf16 elems (16B-multiple, bank-skewed)

typedef __attribute__((ext_vector_type(16))) __bf16 v16bf;
typedef __attribute__((ext_vector_type(8)))  __bf16 v8bf;
typedef __attribute__((ext_vector_type(8)))  float  v8f;

__device__ __forceinline__ v8bf ld8bf(const __bf16* p) {
  union { uint4 u; v8bf b; } cv;
  cv.u = *reinterpret_cast<const uint4*>(p);
  return cv.b;
}
// Build a 16-elem bf16 fragment from two aligned 8-elem LDS chunks.
__device__ __forceinline__ v16bf frag_lds(const __bf16* p0, const __bf16* p1) {
  v8bf lo = ld8bf(p0), hi = ld8bf(p1);
  return __builtin_shufflevector(lo, hi, 0,1,2,3,4,5,6,7,8,9,10,11,12,13,14,15);
}
// Build a fragment straight from global f32 (router only; tiny matrix).
__device__ __forceinline__ v16bf frag_cvt(const float* p0, const float* p1) {
  v16bf r;
#pragma unroll
  for (int i = 0; i < 8; ++i) { r[i] = (__bf16)p0[i]; r[i + 8] = (__bf16)p1[i]; }
  return r;
}
__device__ __forceinline__ v8f wmma_bf16(v16bf a, v16bf b, v8f c) {
  // D = A(16x32) * B(32x16) + C(16x16), f32 accumulate
  return __builtin_amdgcn_wmma_f32_16x16x32_bf16(false, a, false, b, (short)0, c,
                                                 false, false);
}

// CDNA5 async copy: 16B global -> LDS per active lane, tracked on ASYNCcnt.
__device__ __forceinline__ void async_copy_b128(void* lds_ptr, const void* gptr) {
  const unsigned int lds_off = (unsigned int)(unsigned long long)lds_ptr;
  asm volatile("global_load_async_to_lds_b128 %0, %1, off"
               :: "v"(lds_off), "v"(gptr) : "memory");
}
__device__ __forceinline__ void wait_asynccnt0() {
  asm volatile("s_wait_asynccnt 0" ::: "memory");
}

// ---------------------------------------------------------------------------
// Kernel 1: router logits (16x16x1024 per wave) + top-1 argmax + sigmoid score
// ---------------------------------------------------------------------------
__global__ __launch_bounds__(32)
void router_topk_kernel(const float* __restrict__ x, const float* __restrict__ rw,
                        float* __restrict__ logits, int* __restrict__ topk,
                        float* __restrict__ score) {
  const int t0   = blockIdx.x * 16;
  const int lane = threadIdx.x;
  const int ln   = lane & 15, hi = lane >> 4;
  const float* xr = x  + (size_t)(t0 + ln) * H_DIM;  // A row m = lane%16
  const float* wr = rw + (size_t)ln * H_DIM;         // B col n = lane%16
  v8f acc = {};
#pragma unroll 4
  for (int k0 = 0; k0 < H_DIM; k0 += KSTEP) {
    v16bf a = frag_cvt(xr + k0 + 8 * hi, xr + k0 + 16 + 8 * hi);
    v16bf b = frag_cvt(wr + k0 + 8 * hi, wr + k0 + 16 + 8 * hi);
    acc = wmma_bf16(a, b, acc);
  }
#pragma unroll
  for (int j = 0; j < 8; ++j) {
    const int m = hi * 8 + j;                 // C row (lanes 0-15: 0..7, 16-31: 8..15)
    logits[(size_t)(t0 + m) * N_EXP + ln] = acc[j];
    float v = acc[j]; int idx = ln;
    // argmax over 16 experts within each 16-lane half (lowest index on ties)
#pragma unroll
    for (int off = 1; off < 16; off <<= 1) {
      float ov = __shfl_xor(v, off, 32);
      int   oi = __shfl_xor(idx, off, 32);
      if (ov > v || (ov == v && oi < idx)) { v = ov; idx = oi; }
    }
    if (ln == 0) {
      topk[t0 + m]  = idx;
      score[t0 + m] = 1.f / (1.f + __expf(-v));   // sigmoid(top logit)
    }
  }
}

// ---------------------------------------------------------------------------
// Kernel 2: bucket tokens by expert (counts, exclusive offsets, compact list)
// ---------------------------------------------------------------------------
__global__ __launch_bounds__(1024)
void bucket_kernel(const int* __restrict__ topk, int* __restrict__ counts,
                   int* __restrict__ offsets, int* __restrict__ list) {
  __shared__ int s_cnt[N_EXP], s_cur[N_EXP];
  const int tid = threadIdx.x;
  if (tid < N_EXP) s_cnt[tid] = 0;
  __syncthreads();
  for (int t = tid; t < T_TOK; t += 1024) atomicAdd(&s_cnt[topk[t]], 1);
  __syncthreads();
  if (tid == 0) {
    int run = 0;
    for (int e = 0; e < N_EXP; ++e) {
      s_cur[e] = run; offsets[e] = run; counts[e] = s_cnt[e]; run += s_cnt[e];
    }
  }
  __syncthreads();
  for (int t = tid; t < T_TOK; t += 1024) {
    const int slot = atomicAdd(&s_cur[topk[t]], 1);
    list[slot] = t;
  }
}

// ---------------------------------------------------------------------------
// Kernel 3: pass 1 — act = silu(x @ Wg^T) * (x @ Wu^T), bf16 out to workspace.
// grid.z = 17: z<16 routed experts (gathered rows), z==16 shared expert.
// Tile 32M x 128N; each staged weight fragment feeds 2 WMMAs (M sub-tiles).
// ---------------------------------------------------------------------------
__global__ __launch_bounds__(256)
void mlp_pass1_kernel(const float* __restrict__ x,
                      const float* __restrict__ gate_w, const float* __restrict__ up_w,
                      const float* __restrict__ sh_gate, const float* __restrict__ sh_up,
                      const int* __restrict__ counts, const int* __restrict__ offsets,
                      const int* __restrict__ list,
                      __bf16* __restrict__ act_shared, __bf16* __restrict__ act_routed) {
  const int  e     = blockIdx.z;
  const bool shexp = (e == N_EXP);
  const int  count = shexp ? T_TOK : counts[e];
  const int  m0    = blockIdx.y * BM;
  if (m0 >= count) return;                       // expert has fewer tokens
  const int n0 = blockIdx.x * BN;
  const float* Wg = shexp ? sh_gate : gate_w + (size_t)e * I_DIM * H_DIM;
  const float* Wu = shexp ? sh_up   : up_w   + (size_t)e * I_DIM * H_DIM;
  __bf16* act = shexp ? act_shared : act_routed;
  const int row_base = shexp ? 0 : offsets[e];

  __shared__ __align__(16) __bf16 sA[BM][LDSW];
  __shared__ __align__(16) __bf16 sG[BN][LDSW];
  __shared__ __align__(16) __bf16 sU[BN][LDSW];
  __shared__ int s_tok[BM];

  const int tid = threadIdx.x;
  if (tid < BM) {
    const int r  = m0 + tid;
    const int rr = (r < count) ? r : (count - 1);  // clamp (rows guarded at store)
    s_tok[tid] = shexp ? rr : list[row_base + rr];
  }
  __syncthreads();

  const int wv = tid >> 5, lane = tid & 31, ln = lane & 15, hi = lane >> 4;
  v8f g0 = {}, g1 = {}, u0 = {}, u1 = {};

  for (int k0 = 0; k0 < H_DIM; k0 += KSTEP) {
    {   // stage A: 32x32 f32 -> bf16 (float4 / thread, coalesced per row)
      const int idx = tid * 4, r = idx >> 5, c = idx & 31;
      const float4 v = *reinterpret_cast<const float4*>(
          x + (size_t)s_tok[r] * H_DIM + k0 + c);
      sA[r][c] = (__bf16)v.x; sA[r][c+1] = (__bf16)v.y;
      sA[r][c+2] = (__bf16)v.z; sA[r][c+3] = (__bf16)v.w;
    }
#pragma unroll
    for (int i = 0; i < 4; ++i) {   // stage gate/up: 128x32 f32 -> bf16 each
      const int lin = tid + 256 * i;
      const int n = lin >> 3, k4 = (lin & 7) * 4;
      const float4 g = *reinterpret_cast<const float4*>(
          Wg + (size_t)(n0 + n) * H_DIM + k0 + k4);
      sG[n][k4] = (__bf16)g.x; sG[n][k4+1] = (__bf16)g.y;
      sG[n][k4+2] = (__bf16)g.z; sG[n][k4+3] = (__bf16)g.w;
      const float4 u = *reinterpret_cast<const float4*>(
          Wu + (size_t)(n0 + n) * H_DIM + k0 + k4);
      sU[n][k4] = (__bf16)u.x; sU[n][k4+1] = (__bf16)u.y;
      sU[n][k4+2] = (__bf16)u.z; sU[n][k4+3] = (__bf16)u.w;
    }
    if (k0 + KSTEP < H_DIM) {       // pull next weight slab toward L2/L0
      __builtin_prefetch(Wg + (size_t)(n0 + (tid >> 1)) * H_DIM + k0 + KSTEP, 0, 0);
      __builtin_prefetch(Wu + (size_t)(n0 + (tid >> 1)) * H_DIM + k0 + KSTEP, 0, 0);
    }
    __syncthreads();
    const v16bf a0 = frag_lds(&sA[ln][8 * hi], &sA[ln][16 + 8 * hi]);
    const v16bf a1 = frag_lds(&sA[16 + ln][8 * hi], &sA[16 + ln][16 + 8 * hi]);
    const v16bf bg = frag_lds(&sG[wv * 16 + ln][8 * hi], &sG[wv * 16 + ln][16 + 8 * hi]);
    const v16bf bu = frag_lds(&sU[wv * 16 + ln][8 * hi], &sU[wv * 16 + ln][16 + 8 * hi]);
    g0 = wmma_bf16(a0, bg, g0);
    g1 = wmma_bf16(a1, bg, g1);
    u0 = wmma_bf16(a0, bu, u0);
    u1 = wmma_bf16(a1, bu, u1);
    __syncthreads();
  }

  const int col = n0 + wv * 16 + ln;
#pragma unroll
  for (int j = 0; j < 8; ++j) {
    const int mr = hi * 8 + j;
    if (m0 + mr < count) {
      const float g = g0[j], u = u0[j];
      const float v = (g / (1.f + __expf(-g))) * u;   // silu(g) * u
      act[(size_t)(row_base + m0 + mr) * I_DIM + col] = (__bf16)v;
    }
    if (m0 + 16 + mr < count) {
      const float g = g1[j], u = u1[j];
      const float v = (g / (1.f + __expf(-g))) * u;
      act[(size_t)(row_base + m0 + 16 + mr) * I_DIM + col] = (__bf16)v;
    }
  }
}

// ---------------------------------------------------------------------------
// Kernel 4: pass 2 — down-projection. routed==0: out = act_sh @ Wd_sh^T.
// routed==1: out[t] += score[t] * (act_e @ Wd_e^T)  (top-1 => unique writer).
// A-tile (pure bf16 copy) staged via async global->LDS on ASYNCcnt.
// ---------------------------------------------------------------------------
__global__ __launch_bounds__(256)
void mlp_pass2_kernel(const __bf16* __restrict__ act_shared,
                      const __bf16* __restrict__ act_routed,
                      const float* __restrict__ down_w, const float* __restrict__ sh_down,
                      const int* __restrict__ counts, const int* __restrict__ offsets,
                      const int* __restrict__ list, const float* __restrict__ score,
                      float* __restrict__ out, int routed) {
  const int  e     = routed ? blockIdx.z : N_EXP;
  const bool shexp = !routed;
  const int  count = shexp ? T_TOK : counts[e];
  const int  m0    = blockIdx.y * BM;
  if (m0 >= count) return;
  const int n0 = blockIdx.x * BN;                       // over H
  const float*  Wd = shexp ? sh_down : down_w + (size_t)e * H_DIM * I_DIM;
  const __bf16* A  = shexp ? act_shared : act_routed;
  const int row_base = shexp ? 0 : offsets[e];

  __shared__ __align__(16) __bf16 sA[BM][LDSW];
  __shared__ __align__(16) __bf16 sB[BN][LDSW];
  __shared__ int s_row[BM];
  __shared__ int s_dst[BM];

  const int tid = threadIdx.x;
  if (tid < BM) {
    const int r  = m0 + tid;
    const int rr = (r < count) ? r : (count - 1);
    s_row[tid] = row_base + rr;
    s_dst[tid] = shexp ? rr : list[row_base + rr];
  }
  __syncthreads();

  const int wv = tid >> 5, lane = tid & 31, ln = lane & 15, hi = lane >> 4;
  v8f c0 = {}, c1 = {};

  for (int k0 = 0; k0 < I_DIM; k0 += KSTEP) {
    // stage A: 32x32 bf16 tile (2 KB) = 128 x 16B async copies, waves 0-3
    if (tid < 128) {
      const int r = tid >> 2, cc = (tid & 3) * 8;       // 8 bf16 = 16 B per lane
      async_copy_b128(&sA[r][cc], A + (size_t)s_row[r] * I_DIM + k0 + cc);
    }
#pragma unroll
    for (int i = 0; i < 4; ++i) {   // stage down weights: 128x32 f32 -> bf16
      const int lin = tid + 256 * i;
      const int n = lin >> 3, k4 = (lin & 7) * 4;
      const float4 d = *reinterpret_cast<const float4*>(
          Wd + (size_t)(n0 + n) * I_DIM + k0 + k4);
      sB[n][k4] = (__bf16)d.x; sB[n][k4+1] = (__bf16)d.y;
      sB[n][k4+2] = (__bf16)d.z; sB[n][k4+3] = (__bf16)d.w;
    }
    if (k0 + KSTEP < I_DIM)
      __builtin_prefetch(Wd + (size_t)(n0 + (tid >> 1)) * I_DIM + k0 + KSTEP, 0, 0);
    if (tid < 128) wait_asynccnt0();                    // issuing waves drain ASYNCcnt
    __syncthreads();
    const v16bf a0 = frag_lds(&sA[ln][8 * hi], &sA[ln][16 + 8 * hi]);
    const v16bf a1 = frag_lds(&sA[16 + ln][8 * hi], &sA[16 + ln][16 + 8 * hi]);
    const v16bf b  = frag_lds(&sB[wv * 16 + ln][8 * hi], &sB[wv * 16 + ln][16 + 8 * hi]);
    c0 = wmma_bf16(a0, b, c0);
    c1 = wmma_bf16(a1, b, c1);
    __syncthreads();
  }

  const int col = n0 + wv * 16 + ln;
#pragma unroll
  for (int j = 0; j < 8; ++j) {
    const int mr = hi * 8 + j;
    if (m0 + mr < count) {
      const int t = s_dst[mr];
      const size_t o = (size_t)t * H_DIM + col;
      if (shexp) out[o] = c0[j];
      else       out[o] += score[t] * c0[j];   // unique writer per (t, col)
    }
    if (m0 + 16 + mr < count) {
      const int t = s_dst[16 + mr];
      const size_t o = (size_t)t * H_DIM + col;
      if (shexp) out[o] = c1[j];
      else       out[o] += score[t] * c1[j];
    }
  }
}

// ---------------------------------------------------------------------------
extern "C" void kernel_launch(void* const* d_in, const int* in_sizes, int n_in,
                              void* d_out, int out_size, void* d_ws, size_t ws_size,
                              hipStream_t stream) {
  (void)in_sizes; (void)n_in; (void)out_size; (void)ws_size;
  const float* x        = (const float*)d_in[0];   // [1, 2048, 1024]
  const float* router_w = (const float*)d_in[1];   // [16, 1024]
  const float* gate_w   = (const float*)d_in[2];   // [16, 2048, 1024]
  const float* up_w     = (const float*)d_in[3];   // [16, 2048, 1024]
  const float* down_w   = (const float*)d_in[4];   // [16, 1024, 2048]
  const float* sh_gate  = (const float*)d_in[5];   // [2048, 1024]
  const float* sh_up    = (const float*)d_in[6];   // [2048, 1024]
  const float* sh_down  = (const float*)d_in[7];   // [1024, 2048]

  float* out    = (float*)d_out;                   // [T, H]
  float* logits = out + (size_t)T_TOK * H_DIM;     // [T, E]

  char* ws      = (char*)d_ws;
  int*  counts  = (int*)ws;                        // [16]
  int*  offsets = counts + 16;                     // [16]
  int*  topk    = offsets + 16;                    // [T]
  int*  list    = topk + T_TOK;                    // [T] compacted per-expert
  float* score  = (float*)(list + T_TOK);          // [T]
  __bf16* act_shared = (__bf16*)(ws + (1 << 16));              // [T, I] bf16 (8 MB)
  __bf16* act_routed = act_shared + (size_t)T_TOK * I_DIM;     // [T, I] bf16 (8 MB)

  router_topk_kernel<<<T_TOK / 16, 32, 0, stream>>>(x, router_w, logits, topk, score);
  bucket_kernel<<<1, 1024, 0, stream>>>(topk, counts, offsets, list);
  mlp_pass1_kernel<<<dim3(I_DIM / BN, T_TOK / BM, N_EXP + 1), 256, 0, stream>>>(
      x, gate_w, up_w, sh_gate, sh_up, counts, offsets, list, act_shared, act_routed);
  // shared down-proj must complete before routed accumulation (stream order)
  mlp_pass2_kernel<<<dim3(H_DIM / BN, T_TOK / BM, 1), 256, 0, stream>>>(
      act_shared, act_routed, down_w, sh_down, counts, offsets, list, score, out, 0);
  mlp_pass2_kernel<<<dim3(H_DIM / BN, T_TOK / BM, N_EXP), 256, 0, stream>>>(
      act_shared, act_routed, down_w, sh_down, counts, offsets, list, score, out, 1);
}
